// HybridKimiModel_66056597012604
// MI455X (gfx1250) — matrix operand order repaired and checked by reading, not compiled, not run on hardware
//
#include <hip/hip_runtime.h>
#include <hip/hip_bf16.h>

// ---------------------------------------------------------------------------
// Hybrid Kimi model forward (B=1, T=1024, D=1024, H=16, DK=64, L=8, V=32000)
// CDNA5 / gfx1250: wave32, WMMA bf16 16x16x32, double-buffered LDS GEMMs,
// TDM tensor_load_to_lds + async global<->LDS ops for the embedding gather.
// ---------------------------------------------------------------------------

#define T_LEN 1024
#define DMODEL 1024
#define NHEAD 16
#define HDIM 64
#define VOCAB 32000

typedef __bf16 bf16_t;
typedef bf16_t bf16x8  __attribute__((ext_vector_type(8)));
typedef bf16_t bf16x16 __attribute__((ext_vector_type(16)));
typedef float  f32x8   __attribute__((ext_vector_type(8)));
typedef unsigned int u32x4 __attribute__((ext_vector_type(4)));
typedef int i32x4 __attribute__((ext_vector_type(4)));
typedef int i32x8 __attribute__((ext_vector_type(8)));

union BFrag { bf16x16 v; bf16x8 h[2]; };

// Pack two fp32 -> two bf16 in one VALU op (v_cvt_pk_bf16_f32, RNE).
static __device__ inline void st_bf2(bf16_t* dst, float a, float b) {
    unsigned pk;
    asm("v_cvt_pk_bf16_f32 %0, %1, %2" : "=v"(pk) : "v"(a), "v"(b));
    *(unsigned*)dst = pk;
}

// ---------------------------------------------------------------------------
// Embedding gather: x[t,:] = emb[idx[t],:].
// Global -> LDS via the Tensor Data Mover (one 1-row tile descriptor per
// block, issued by wave 0, TENSORcnt), LDS -> global via async store
// (ASYNCcnt). Exercises the gfx1250 TDM + async paths.
// ---------------------------------------------------------------------------
__global__ void embed_kernel(const int* __restrict__ idx,
                             const float* __restrict__ emb,
                             float* __restrict__ x) {
    __shared__ alignas(16) float buf[DMODEL];
    int t = blockIdx.x;
    int row = idx[t];                       // uniform -> SGPR

    if (threadIdx.x < 32) {                 // one wave issues the TDM op
        unsigned long long gaddr =
            (unsigned long long)(const void*)(emb + (size_t)row * DMODEL);
        unsigned lds_base = (unsigned)(unsigned long long)(void*)&buf[0];

        // D# group 0: count=1 | lds_addr | global_addr[56:0] | type=2
        u32x4 g0;
        g0[0] = 1u;
        g0[1] = lds_base;
        g0[2] = (unsigned)(gaddr & 0xFFFFFFFFu);
        g0[3] = (unsigned)((gaddr >> 32) & 0x01FFFFFFu) | (2u << 30);

        // D# group 1: data_size=4B; tensor_dim0=1024; tensor_dim1=1;
        //             tile_dim0=1024; tile_dim1=1; tensor_dim0_stride=1024
        i32x8 g1;
        g1[0] = (int)(2u << 16);                                  // data_size
        g1[1] = (int)(((unsigned)DMODEL & 0xFFFFu) << 16);        // dim0 lo16
        g1[2] = (int)(((unsigned)DMODEL >> 16) | (1u << 16));     // dim0 hi | dim1 lo16=1
        g1[3] = (int)((unsigned)DMODEL << 16);                    // dim1 hi=0 | tile_dim0
        g1[4] = 1;                                                // tile_dim1=1
        g1[5] = DMODEL;                                           // dim0_stride lo32
        g1[6] = 0;
        g1[7] = 0;

        i32x4 z4 = {0, 0, 0, 0};
#if __clang_major__ >= 23
        i32x8 z8 = {0, 0, 0, 0, 0, 0, 0, 0};
        __builtin_amdgcn_tensor_load_to_lds(g0, g1, z4, z4, z8, 0);
#else
        __builtin_amdgcn_tensor_load_to_lds(g0, g1, z4, z4, 0);
#endif
        __builtin_amdgcn_s_wait_tensorcnt(0);
    }
    __syncthreads();

    float* dst = x + (size_t)t * DMODEL + threadIdx.x * 4;
    unsigned lds = (unsigned)(unsigned long long)(void*)&buf[threadIdx.x * 4];
    asm volatile("global_store_async_from_lds_b128 %0, %1, off"
                 :: "v"(dst), "v"(lds) : "memory");
    asm volatile("s_wait_asynccnt 0x0" ::: "memory");
}

// ---------------------------------------------------------------------------
// Tiled WMMA GEMM: C[M,N] = A[M,K] @ B  (B is [K,N] or, if TRANSB, [N,K]).
// Block = 256 threads (8 waves). Block tile: 32(M) x 128(N), K step 32.
// Double-buffered LDS (one barrier per K step); each wave does 2 WMMAs
// (A fragment reused across two N sub-tiles).
// fp32 global -> bf16 LDS -> v_wmma_f32_16x16x32_bf16, fp32 accum.
// ---------------------------------------------------------------------------
#define ASTR 40   // halves; 80B row stride: 16B aligned, bank-conflict-free
#define BSTR 40

template <bool TRANSB>
__global__ void gemm_bf16_wmma(const float* __restrict__ A,
                               const float* __restrict__ B,
                               float* __restrict__ C,
                               int M, int N, int K) {
    __shared__ alignas(16) bf16_t As[2][32 * ASTR];
    __shared__ alignas(16) bf16_t Bs[2][128 * BSTR];

    int tid  = threadIdx.x;
    int lane = tid & 31;
    int w    = tid >> 5;            // 0..7
    int m0   = blockIdx.x * 32;
    int n0   = blockIdx.y * 128;
    int wm   = (w >> 2) * 16;       // wave M sub-tile (0/16)
    int wn0  = (w & 3) * 32;        // wave N base (0/32/64/96), 2 subtiles

    f32x8 acc0 = {0.f,0.f,0.f,0.f,0.f,0.f,0.f,0.f};
    f32x8 acc1 = {0.f,0.f,0.f,0.f,0.f,0.f,0.f,0.f};

    // cooperative-load index maps
    int ar = tid >> 3;              // A: row 0..31
    int ac = (tid & 7) * 4;         // A: 4 contiguous K
    int bn, bk;
    if (TRANSB) { bn = tid >> 1;   bk = (tid & 1) * 16; }   // row-major [N,K]
    else        { bn = tid & 127;  bk = (tid >> 7) * 16; }  // column gather of [K,N]

    float4 aReg;
    float  bReg[16];

    auto loadG = [&](int kc) {
        const float* ap = A + (size_t)(m0 + ar) * K + kc + ac;
        if (kc + 32 < K) __builtin_prefetch(ap + 32, 0, 1);   // global_prefetch_b8
        aReg = *(const float4*)ap;
        int gn = n0 + bn;
        if (gn < N) {
            if (TRANSB) {
                const float* bp = B + (size_t)gn * K + kc + bk;
                if (kc + 32 < K) __builtin_prefetch(bp + 32, 0, 1);
                float4 b0 = *(const float4*)(bp + 0);
                float4 b1 = *(const float4*)(bp + 4);
                float4 b2 = *(const float4*)(bp + 8);
                float4 b3 = *(const float4*)(bp + 12);
                bReg[0]=b0.x; bReg[1]=b0.y; bReg[2]=b0.z; bReg[3]=b0.w;
                bReg[4]=b1.x; bReg[5]=b1.y; bReg[6]=b1.z; bReg[7]=b1.w;
                bReg[8]=b2.x; bReg[9]=b2.y; bReg[10]=b2.z; bReg[11]=b2.w;
                bReg[12]=b3.x; bReg[13]=b3.y; bReg[14]=b3.z; bReg[15]=b3.w;
            } else {
#pragma unroll
                for (int i = 0; i < 16; ++i)
                    bReg[i] = B[(size_t)(kc + bk + i) * N + gn];
            }
        } else {
#pragma unroll
            for (int i = 0; i < 16; ++i) bReg[i] = 0.f;
        }
    };

    auto storeL = [&](int s) {
        bf16_t* ad = &As[s][ar * ASTR + ac];
        st_bf2(ad + 0, aReg.x, aReg.y);
        st_bf2(ad + 2, aReg.z, aReg.w);
        bf16_t* bd = &Bs[s][bn * BSTR + bk];
#pragma unroll
        for (int i = 0; i < 8; ++i)
            st_bf2(bd + 2 * i, bReg[2 * i], bReg[2 * i + 1]);
    };

    loadG(0);
    storeL(0);
    __syncthreads();

    int nk = K / 32;
    for (int it = 0; it < nk; ++it) {
        int cur = it & 1;
        bool more = (it + 1) < nk;
        if (more) loadG((it + 1) * 32);   // global loads overlap WMMA below

        // ---- fragments per documented wave32 layouts ----
        // A 16x32: lane<16 -> K {0..7,16..23}; lane>=16 -> K {8..15,24..31}
        BFrag a, b0, b1;
        int am  = lane & 15;
        int akb = (lane >> 4) * 8;
        const bf16_t* Ab = &As[cur][(wm + am) * ASTR];
        a.h[0] = *(const bf16x8*)(Ab + akb);
        a.h[1] = *(const bf16x8*)(Ab + 16 + akb);
        // B 32x16: lane<16 -> (N=lane, K {0..7,16..23}); lane>=16 -> (N=lane-16, K {8..15,24..31})
        int bnn = lane & 15;
        int bkb = (lane >> 4) * 8;
        const bf16_t* Bb0 = &Bs[cur][(wn0 + bnn) * BSTR];
        b0.h[0] = *(const bf16x8*)(Bb0 + bkb);
        b0.h[1] = *(const bf16x8*)(Bb0 + 16 + bkb);
        const bf16_t* Bb1 = &Bs[cur][(wn0 + 16 + bnn) * BSTR];
        b1.h[0] = *(const bf16x8*)(Bb1 + bkb);
        b1.h[1] = *(const bf16x8*)(Bb1 + 16 + bkb);

        acc0 = __builtin_amdgcn_wmma_f32_16x16x32_bf16(
            false, a.v, false, b0.v, (short)0, acc0, false, false);
        acc1 = __builtin_amdgcn_wmma_f32_16x16x32_bf16(
            false, a.v, false, b1.v, (short)0, acc1, false, false);

        if (more) storeL(cur ^ 1);
        __syncthreads();
    }

    // ---- C 16x16 f32 layout: M = r + 8*(lane>>4), N = lane&15 ----
    int cm  = m0 + wm + (lane >> 4) * 8;
    int cn0 = n0 + wn0 + (lane & 15);
    if (cn0 < N) {
#pragma unroll
        for (int r = 0; r < 8; ++r) C[(size_t)(cm + r) * N + cn0] = acc0[r];
    }
    int cn1 = cn0 + 16;
    if (cn1 < N) {
#pragma unroll
        for (int r = 0; r < 8; ++r) C[(size_t)(cm + r) * N + cn1] = acc1[r];
    }
}

// ---------------------------------------------------------------------------
// Per-head L2 normalize: x viewed as [T*H, 64]; one wave32 per row.
// ---------------------------------------------------------------------------
__global__ void l2norm_kernel(float* __restrict__ x) {
    int tid  = threadIdx.x;
    int lane = tid & 31;
    int w = blockIdx.x * (blockDim.x >> 5) + (tid >> 5);
    float* p = x + (size_t)w * 64;
    float a = p[lane], b = p[lane + 32];
    float s = a * a + b * b;
#pragma unroll
    for (int off = 16; off; off >>= 1) s += __shfl_xor(s, off, 32);
    float r = rsqrtf(s + 1e-6f);
    p[lane] = a * r;
    p[lane + 32] = b * r;
}

// ---------------------------------------------------------------------------
// Elementwise sigmoid
// ---------------------------------------------------------------------------
__global__ void sigmoid_kernel(float* __restrict__ x, int n) {
    int i = blockIdx.x * blockDim.x + threadIdx.x;
    if (i < n) x[i] = 1.f / (1.f + __expf(-x[i]));
}

// ---------------------------------------------------------------------------
// KDA gated delta-rule scan. One block per head (16 blocks, 64 threads).
// Thread tid owns state column S[:, tid] (64 VGPRs).
//   S = g*S;  kS = k^T S;  S += beta*k*(v - kS);  o = q^T S
// ---------------------------------------------------------------------------
__global__ void kda_scan_kernel(const float* __restrict__ q,
                                const float* __restrict__ k,
                                const float* __restrict__ v,
                                const float* __restrict__ beta,
                                const float* __restrict__ g,
                                float* __restrict__ o) {
    int h = blockIdx.x;
    int tid = threadIdx.x;                 // 0..63 == dv
    __shared__ float qs[64], ks[64];
    float S[64];
#pragma unroll
    for (int i = 0; i < 64; ++i) S[i] = 0.f;

    for (int t = 0; t < T_LEN; ++t) {
        int base = t * DMODEL + h * HDIM;
        qs[tid] = q[base + tid];
        ks[tid] = k[base + tid];
        float vt = v[base + tid];
        float bt = beta[t * NHEAD + h];
        float gt = g[t * NHEAD + h];
        __syncthreads();

        float kS = 0.f;
#pragma unroll
        for (int i = 0; i < 64; ++i) { S[i] *= gt; kS = fmaf(ks[i], S[i], kS); }
        float delta = bt * (vt - kS);
        float ot = 0.f;
#pragma unroll
        for (int i = 0; i < 64; ++i) { S[i] = fmaf(ks[i], delta, S[i]); ot = fmaf(qs[i], S[i], ot); }
        o[base + tid] = ot;
        __syncthreads();
    }
}

// ---------------------------------------------------------------------------
// Causal softmax attention, online softmax, one wave32 per (head, query row).
// q,k,v,o: [T, D] with head offset h*64; scale = 1/sqrt(64).
// ---------------------------------------------------------------------------
__global__ void attn_kernel(const float* __restrict__ q,
                            const float* __restrict__ k,
                            const float* __restrict__ v,
                            float* __restrict__ o) {
    int tid  = threadIdx.x;
    int lane = tid & 31;
    int w = blockIdx.x * (blockDim.x >> 5) + (tid >> 5);   // 0..H*T-1
    int h  = w >> 10;        // w / T
    int qi = w & (T_LEN - 1);

    const float* qp = q + (size_t)qi * DMODEL + h * HDIM;
    float q0 = qp[lane] * 0.125f;
    float q1 = qp[lane + 32] * 0.125f;

    float mmax = -3.4e38f, l = 0.f, a0 = 0.f, a1 = 0.f;
    for (int j = 0; j <= qi; ++j) {
        const float* kp = k + (size_t)j * DMODEL + h * HDIM;
        float s = q0 * kp[lane] + q1 * kp[lane + 32];
#pragma unroll
        for (int off = 16; off; off >>= 1) s += __shfl_xor(s, off, 32);
        float mn   = fmaxf(mmax, s);
        float corr = __expf(mmax - mn);
        float p    = __expf(s - mn);
        const float* vp = v + (size_t)j * DMODEL + h * HDIM;
        l  = l * corr + p;
        a0 = a0 * corr + p * vp[lane];
        a1 = a1 * corr + p * vp[lane + 32];
        mmax = mn;
    }
    float inv = 1.f / l;
    float* op = o + (size_t)qi * DMODEL + h * HDIM;
    op[lane] = a0 * inv;
    op[lane + 32] = a1 * inv;
}

// ---------------------------------------------------------------------------
// Final LayerNorm over D=1024, one block per row.
// ---------------------------------------------------------------------------
__global__ void layernorm_kernel(const float* __restrict__ x,
                                 const float* __restrict__ gw,
                                 const float* __restrict__ bw,
                                 float* __restrict__ y) {
    __shared__ float s1[256], s2[256];
    int t = blockIdx.x, tid = threadIdx.x;
    const float* xr = x + (size_t)t * DMODEL;
    float a = 0.f, b = 0.f;
    for (int i = tid; i < DMODEL; i += 256) { float v = xr[i]; a += v; b += v * v; }
    s1[tid] = a; s2[tid] = b;
    __syncthreads();
    for (int s = 128; s; s >>= 1) {
        if (tid < s) { s1[tid] += s1[tid + s]; s2[tid] += s2[tid + s]; }
        __syncthreads();
    }
    float mu   = s1[0] * (1.f / DMODEL);
    float var  = s2[0] * (1.f / DMODEL) - mu * mu;
    float rstd = rsqrtf(var + 1e-5f);
    for (int i = tid; i < DMODEL; i += 256)
        y[(size_t)t * DMODEL + i] = (xr[i] - mu) * rstd * gw[i] + bw[i];
}

// ---------------------------------------------------------------------------
// Host launcher
// ---------------------------------------------------------------------------
static void launch_gemm(const float* A, const float* B, float* C,
                        int M, int N, int K, bool transB, hipStream_t s) {
    dim3 grid(M / 32, (N + 127) / 128);  // grid.x fastest => same N-tile blocks
    dim3 block(256);                     // run consecutively -> weights hit L2
    if (transB) gemm_bf16_wmma<true><<<grid, block, 0, s>>>(A, B, C, M, N, K);
    else        gemm_bf16_wmma<false><<<grid, block, 0, s>>>(A, B, C, M, N, K);
}

extern "C" void kernel_launch(void* const* d_in, const int* in_sizes, int n_in,
                              void* d_out, int out_size, void* d_ws, size_t ws_size,
                              hipStream_t stream) {
    // ---- map inputs (canonical: kda {wq,wk,wv,wo,wb,wg}, attn {wq,wk,wv,wo}) ----
    const int* idx;
    const float *emb, *ln_g, *ln_b;
    const float* kda_w[6][6];
    const float* attn_w[2][4];

    if (n_in >= 2 && in_sizes[1] == VOCAB * DMODEL) {
        // insertion-order pytree flatten: idx, emb, kda[...], attn[...], ln_g, ln_b
        int p = 0;
        idx = (const int*)d_in[p++];
        emb = (const float*)d_in[p++];
        for (int l = 0; l < 6; ++l)
            for (int j = 0; j < 6; ++j) kda_w[l][j] = (const float*)d_in[p++];
        for (int l = 0; l < 2; ++l)
            for (int j = 0; j < 4; ++j) attn_w[l][j] = (const float*)d_in[p++];
        ln_g = (const float*)d_in[p++];
        ln_b = (const float*)d_in[p++];
    } else {
        // sorted-key flatten: idx, attn{wk,wo,wq,wv}x2, emb, kda{wb,wg,wk,wo,wq,wv}x6, ln_b, ln_g
        int p = 0;
        idx = (const int*)d_in[p++];
        for (int l = 0; l < 2; ++l) {
            attn_w[l][1] = (const float*)d_in[p++];   // wk
            attn_w[l][3] = (const float*)d_in[p++];   // wo
            attn_w[l][0] = (const float*)d_in[p++];   // wq
            attn_w[l][2] = (const float*)d_in[p++];   // wv
        }
        emb = (const float*)d_in[p++];
        for (int l = 0; l < 6; ++l) {
            kda_w[l][4] = (const float*)d_in[p++];    // wb
            kda_w[l][5] = (const float*)d_in[p++];    // wg
            kda_w[l][1] = (const float*)d_in[p++];    // wk
            kda_w[l][3] = (const float*)d_in[p++];    // wo
            kda_w[l][0] = (const float*)d_in[p++];    // wq
            kda_w[l][2] = (const float*)d_in[p++];    // wv
        }
        ln_b = (const float*)d_in[p++];
        ln_g = (const float*)d_in[p++];
    }

    // ---- workspace carve (fp32) ----
    const size_t MAT = (size_t)T_LEN * DMODEL;   // 1M floats
    float* ws   = (float*)d_ws;
    float* x    = ws;
    float* qb   = ws + 1 * MAT;
    float* kb   = ws + 2 * MAT;
    float* vb   = ws + 3 * MAT;
    float* ob   = ws + 4 * MAT;
    float* xln  = ws + 5 * MAT;
    float* beb  = ws + 6 * MAT;
    float* gbuf = beb + (size_t)T_LEN * NHEAD;

    // ---- embedding (TDM load + async store staging) ----
    embed_kernel<<<T_LEN, 256, 0, stream>>>(idx, emb, x);

    // ---- layers ----
    int kda_i = 0, attn_i = 0;
    for (int i = 0; i < 8; ++i) {
        if ((i + 1) % 4 == 0) {
            const float* const* p = attn_w[attn_i++];
            launch_gemm(x, p[0], qb, T_LEN, DMODEL, DMODEL, false, stream);
            launch_gemm(x, p[1], kb, T_LEN, DMODEL, DMODEL, false, stream);
            launch_gemm(x, p[2], vb, T_LEN, DMODEL, DMODEL, false, stream);
            attn_kernel<<<(NHEAD * T_LEN) / 8, 256, 0, stream>>>(qb, kb, vb, ob);
            launch_gemm(ob, p[3], x, T_LEN, DMODEL, DMODEL, false, stream);
        } else {
            const float* const* p = kda_w[kda_i++];
            launch_gemm(x, p[0], qb, T_LEN, DMODEL, DMODEL, false, stream);
            launch_gemm(x, p[1], kb, T_LEN, DMODEL, DMODEL, false, stream);
            launch_gemm(x, p[2], vb, T_LEN, DMODEL, DMODEL, false, stream);
            launch_gemm(x, p[4], beb, T_LEN, NHEAD, DMODEL, false, stream);
            launch_gemm(x, p[5], gbuf, T_LEN, NHEAD, DMODEL, false, stream);
            l2norm_kernel<<<(T_LEN * NHEAD) / 8, 256, 0, stream>>>(qb);
            l2norm_kernel<<<(T_LEN * NHEAD) / 8, 256, 0, stream>>>(kb);
            sigmoid_kernel<<<(T_LEN * NHEAD + 255) / 256, 256, 0, stream>>>(beb, T_LEN * NHEAD);
            sigmoid_kernel<<<(T_LEN * NHEAD + 255) / 256, 256, 0, stream>>>(gbuf, T_LEN * NHEAD);
            kda_scan_kernel<<<NHEAD, 64, 0, stream>>>(qb, kb, vb, beb, gbuf, ob);
            launch_gemm(ob, p[3], x, T_LEN, DMODEL, DMODEL, false, stream);
        }
    }

    // ---- final LN + tied head (x_ln @ emb^T) ----
    layernorm_kernel<<<T_LEN, 256, 0, stream>>>(x, ln_g, ln_b, xln);
    launch_gemm(xln, emb, (float*)d_out, T_LEN, VOCAB, DMODEL, true, stream);
}
